// RKT_75977971466529
// MI455X (gfx1250) — compile-verified, matrix operand order checked by reading, not run on hardware
//
#include <hip/hip_runtime.h>
#include <math.h>

typedef _Float16 f16;
typedef __attribute__((ext_vector_type(16))) _Float16 v16h;
typedef __attribute__((ext_vector_type(8)))  _Float16 v8h;
typedef __attribute__((ext_vector_type(4)))  _Float16 v4h;
typedef __attribute__((ext_vector_type(8)))  float    v8f;

#define BB 16
#define LL 1024
#define EE 256
#define NTOK (BB * LL)

// ---- fragment loader ---------------------------------------------------------
// Loads a 16-wide-by-32-deep fragment from a row-major [rows][ld] f16 matrix.
// lane&15 selects the row (M for A-fragments, N for B-fragments); the 16 halves
// cover the 32-deep K dim per ISA 7.12.2 (lanes0-15: k 0..7,16..23 ; lanes16-31:
// k 8..15,24..31).  Requires (row0*ld + k0) to be 8-half aligned -> 2x b128.
__device__ inline v16h load_frag(const f16* base, int ld, int row0, int k0) {
  int lane = threadIdx.x & 31;
  const f16* p = base + (long)(row0 + (lane & 15)) * ld + k0 + ((lane >> 4) << 3);
  v8h lo = *(const v8h*)p;
  v8h hi = *(const v8h*)(p + 16);
  return __builtin_shufflevector(lo, hi, 0, 1, 2, 3, 4, 5, 6, 7,
                                 8, 9, 10, 11, 12, 13, 14, 15);
}

__device__ inline v8f wmma_f16(v16h a, v16h b, v8f c) {
  return __builtin_amdgcn_wmma_f32_16x16x32_f16(false, a, false, b, (short)0, c, false, false);
}

// ---- prep kernels ------------------------------------------------------------
__global__ void cvt_f16(const float* __restrict__ W, long n, f16* __restrict__ D) {
  long idx = (long)blockIdx.x * 256 + threadIdx.x;
  if (idx < n) D[idx] = (f16)W[idx];
}

// Ain[t] = [e*r , e*(1-r)] (512 f16), Qin[t] = item_embeds[nxt[t]] (256 f16)
__global__ void gather_build(const int* __restrict__ q, const int* __restrict__ r,
                             const int* __restrict__ nxt, const float* __restrict__ emb,
                             f16* __restrict__ Ain, f16* __restrict__ Qin) {
  int t = blockIdx.x, c = threadIdx.x;
  float rf = (float)r[t];
  const float* e  = emb + (long)q[t]   * EE;
  const float* qe = emb + (long)nxt[t] * EE;
  float ev = e[c];
  Ain[(long)t * 512 + c]       = (f16)(ev * rf);
  Ain[(long)t * 512 + 256 + c] = (f16)(ev * (1.0f - rf));
  Qin[(long)t * 256 + c]       = (f16)qe[c];
}

// ---- generic wave-tile WMMA GEMM ---------------------------------------------
// D = act((A x W^T + bias) * scale).  A row-major [M][Kdim] (lda=Kdim),
// W row-major [N][Kdim] (ldb=Kdim) -- used directly as the B operand.
// tr_out=0: D row-major [M][N] (ldd=N).  tr_out=1: D^T [N][Mtot] (ldd=Mtot),
// stored as one v8h per lane.
__global__ void gemm_f16(const f16* __restrict__ A, int lda,
                         const f16* __restrict__ W, int ldb,
                         const float* __restrict__ bias,
                         f16* __restrict__ D, int ldd,
                         int Mt, int Nt, int Kdim, float scale,
                         int do_relu, int tr_out) {
  int wave = threadIdx.x >> 5;
  int tile = blockIdx.x * (blockDim.x >> 5) + wave;
  if (tile >= Mt * Nt) return;                 // wave-uniform
  int mt = tile / Nt, nt = tile % Nt;
  int row0 = mt * 16, n0 = nt * 16;
  v8f acc = {};
  for (int k0 = 0; k0 < Kdim; k0 += 32) {
    v16h a = load_frag(A, lda, row0, k0);
    v16h b = load_frag(W, ldb, n0, k0);
    acc = wmma_f16(a, b, acc);
  }
  int lane = threadIdx.x & 31;
  int n = n0 + (lane & 15);
  int mb = row0 + ((lane >> 4) << 3);
  float bv = bias ? bias[n] : 0.0f;
  if (tr_out) {
    v8h h;
#pragma unroll
    for (int j = 0; j < 8; ++j) h[j] = (f16)((acc[j] + bv) * scale);
    *(v8h*)(D + (long)n * ldd + mb) = h;
  } else {
#pragma unroll
    for (int j = 0; j < 8; ++j) {
      float v = (acc[j] + bv) * scale;
      if (do_relu) v = fmaxf(v, 0.0f);
      D[(long)(mb + j) * ldd + n] = (f16)v;
    }
  }
}

// ---- fused attention ---------------------------------------------------------
// grid = BB*64 blocks x 256 threads; one block per (b, 16-row strip).
// dyn LDS (128 KB):
//   S  f32 [16][1024] @ 0       (scores)
//   TP f16 [16][1024] @ 65536   (exp(-|ts|) then overwritten with p)
//   R  f16 [16][1024] @ 98304   (masked rel values; OT/RED overlay after use)
#define SM_S  0
#define SM_TP 65536
#define SM_R  98304
#define SM_BYTES 131072

__global__ void attn_fused(const f16* __restrict__ Q, const f16* __restrict__ Km,
                           const f16* __restrict__ Vt,
                           const float* __restrict__ rel, const float* __restrict__ ts,
                           const float* __restrict__ W_out, const float* __restrict__ b_out,
                           const float* __restrict__ l1p, const float* __restrict__ l2p,
                           float* __restrict__ logits, float* __restrict__ pout) {
  extern __shared__ char smem[];
  float* S  = (float*)(smem + SM_S);
  f16*   TP = (f16*)(smem + SM_TP);
  f16*   R  = (f16*)(smem + SM_R);
  float* OT  = (float*)(smem + SM_R);           // overlays R after phase B
  float* RED = (float*)(smem + SM_R + 16384);

  int b  = blockIdx.x >> 6;
  int it = blockIdx.x & 63;
  int i0 = it * 16;
  long tokBase = (long)b * LL;
  int wave = threadIdx.x >> 5;
  int lane = threadIdx.x & 31;
  float l1 = l1p[0], l2 = l2p[0];

  const f16* Qb = Q  + tokBase * EE;
  const f16* Kb = Km + tokBase * EE;

  // warm L2 with the rel/ts rows phase B will stream (overlaps WMMA phase)
  {
    const float* pfr = rel + (tokBase + i0) * LL;
    const float* pft = ts  + (tokBase + i0) * LL;
    for (int t = threadIdx.x; t < 1024; t += 256) {   // 16 rows * 4KB = 1024 lines
      __builtin_prefetch(pfr + t * 16, 0, 3);
      __builtin_prefetch(pft + t * 16, 0, 3);
    }
  }

  // Phase A: S[16][1024] = (Q strip, pre-scaled 1/sqrt(E)) x K^T
  {
    v16h afr[8];
#pragma unroll
    for (int kk = 0; kk < 8; ++kk) afr[kk] = load_frag(Qb, EE, i0, kk * 32);
    for (int jt = wave; jt < 64; jt += 8) {
      v8f acc = {};
#pragma unroll
      for (int kk = 0; kk < 8; ++kk) {
        v16h bf = load_frag(Kb, EE, jt * 16, kk * 32);  // lane = key token
        acc = wmma_f16(afr[kk], bf, acc);
      }
      int n = jt * 16 + (lane & 15);
      int mb = (lane >> 4) << 3;
#pragma unroll
      for (int j = 0; j < 8; ++j) S[(mb + j) * LL + n] = acc[j];
    }
  }
  __syncthreads();

  // Phase B: per-row (wave per row, 2 rows each): single global pass stages
  // t=exp(-|ts|) and masked rel in LDS f16; then sums; then blended p.
  for (int rr = wave; rr < 16; rr += 8) {
    int i = i0 + rr;
    const float4* ts4  = (const float4*)(ts  + (tokBase + i) * LL);
    const float4* rel4 = (const float4*)(rel + (tokBase + i) * LL);
    const float4* S4   = (const float4*)(S + (long)rr * LL);
    f16* Trow = TP + (long)rr * LL;
    f16* Rrow = R  + (long)rr * LL;

    float m1 = -1e30f, m2 = -1e30f, m3 = -1e30f;
    for (int g = lane; g < 256; g += 32) {
      float4 tv = ts4[g], rv = rel4[g], sv = S4[g];
      float ta[4] = {tv.x, tv.y, tv.z, tv.w};
      float ra[4] = {rv.x, rv.y, rv.z, rv.w};
      float sa[4] = {sv.x, sv.y, sv.z, sv.w};
      v4h th, rh;
#pragma unroll
      for (int c = 0; c < 4; ++c) {
        int j = g * 4 + c;
        float t = __expf(-fabsf(ta[c]));
        th[c] = (f16)t;
        float v = (j > i) ? ra[c] : 0.0f;
        v = (v == 0.0f) ? -10000.0f : v;
        rh[c] = (f16)v;
        m3 = fmaxf(m3, v);
        if (j <= i) { m1 = fmaxf(m1, sa[c]); m2 = fmaxf(m2, t); }
      }
      *(v4h*)(Trow + g * 4) = th;
      *(v4h*)(Rrow + g * 4) = rh;
    }
    for (int o = 16; o > 0; o >>= 1) {
      m1 = fmaxf(m1, __shfl_xor(m1, o));
      m2 = fmaxf(m2, __shfl_xor(m2, o));
      m3 = fmaxf(m3, __shfl_xor(m3, o));
    }
    float s1 = 0.f, s2 = 0.f, s3 = 0.f;
    for (int g = lane; g < 256; g += 32) {
      float4 sv = S4[g];
      float sa[4] = {sv.x, sv.y, sv.z, sv.w};
      v4h th = *(v4h*)(Trow + g * 4);
      v4h rh = *(v4h*)(Rrow + g * 4);
#pragma unroll
      for (int c = 0; c < 4; ++c) {
        int j = g * 4 + c;
        if (j <= i) {
          s1 += __expf(sa[c] - m1);
          s2 += __expf((float)th[c] - m2);
        }
        s3 += __expf((float)rh[c] - m3);
      }
    }
    for (int o = 16; o > 0; o >>= 1) {
      s1 += __shfl_xor(s1, o);
      s2 += __shfl_xor(s2, o);
      s3 += __shfl_xor(s3, o);
    }
    float i1 = 1.f / s1, i2 = 1.f / s2, i3 = 1.f / s3;
    float cp = (1.f - l1) * (1.f - l2), ct = (1.f - l1) * l2;
    float4* prow = (float4*)(pout + (tokBase + i) * LL);
    for (int g = lane; g < 256; g += 32) {
      float4 sv = S4[g];
      float sa[4] = {sv.x, sv.y, sv.z, sv.w};
      v4h th = *(v4h*)(Trow + g * 4);
      v4h rh = *(v4h*)(Rrow + g * 4);
      float pa[4];
      v4h ph;
#pragma unroll
      for (int c = 0; c < 4; ++c) {
        int j = g * 4 + c;
        float prob = 0.f, time = 0.f;
        if (j <= i) {
          prob = __expf(sa[c] - m1) * i1;
          time = __expf((float)th[c] - m2) * i2;
        }
        float ra = __expf((float)rh[c] - m3) * i3;
        float p = cp * prob + ct * time + l1 * ra;
        pa[c] = p;
        ph[c] = (f16)p;
      }
      prow[g] = make_float4(pa[0], pa[1], pa[2], pa[3]);
      *(v4h*)(Trow + g * 4) = ph;               // p (f16) overwrites T buffer
    }
  }
  __syncthreads();

  // Phase C: OT[16][256] = p (LDS f16) x V   (Vt is [EE][NTOK], contiguous k)
  for (int nt = wave; nt < 16; nt += 8) {
    v8f acc = {};
    for (int k0 = 0; k0 < LL; k0 += 32) {
      v16h a  = load_frag(TP, LL, 0, k0);                  // lane = p row
      v16h bf = load_frag(Vt + tokBase, NTOK, nt * 16, k0); // lane = embed col
      acc = wmma_f16(a, bf, acc);
    }
    int n = nt * 16 + (lane & 15);
    int mb = (lane >> 4) << 3;
#pragma unroll
    for (int j = 0; j < 8; ++j) OT[(mb + j) * EE + n] = acc[j];
  }
  __syncthreads();

  // Phase D: logits = OT x W_out^T + b_out
  {
    int rr = threadIdx.x >> 4;
    int cb = threadIdx.x & 15;
    float s = 0.f;
#pragma unroll
    for (int k = 0; k < 16; ++k) s += OT[rr * EE + cb + k * 16] * W_out[cb + k * 16];
    RED[threadIdx.x] = s;
    __syncthreads();
    if (cb == 0) {
      float t = 0.f;
#pragma unroll
      for (int k = 0; k < 16; ++k) t += RED[rr * 16 + k];
      logits[tokBase + i0 + rr] = t + b_out[0];
    }
  }
}

// ---- host --------------------------------------------------------------------
extern "C" void kernel_launch(void* const* d_in, const int* in_sizes, int n_in,
                              void* d_out, int out_size, void* d_ws, size_t ws_size,
                              hipStream_t stream) {
  const int*   q    = (const int*)d_in[0];
  const int*   r    = (const int*)d_in[1];
  const int*   nxt  = (const int*)d_in[2];
  const float* rel  = (const float*)d_in[3];
  const float* ts   = (const float*)d_in[4];
  const float* emb  = (const float*)d_in[5];
  const float* W_in = (const float*)d_in[6];
  const float* b_in = (const float*)d_in[7];
  const float* Wq   = (const float*)d_in[8];
  const float* bq   = (const float*)d_in[9];
  const float* Wk   = (const float*)d_in[10];
  const float* bk   = (const float*)d_in[11];
  const float* Wv   = (const float*)d_in[12];
  const float* bv   = (const float*)d_in[13];
  const float* Wout = (const float*)d_in[14];
  const float* bout = (const float*)d_in[15];
  const float* l1   = (const float*)d_in[16];
  const float* l2   = (const float*)d_in[17];

  char* ws = (char*)d_ws;
  size_t off = 0;
  auto alloc = [&](size_t bytes) -> char* {
    char* p = ws + off;
    off += (bytes + 255) & ~(size_t)255;
    return p;
  };
  f16* Ain  = (f16*)alloc((size_t)NTOK * 512 * 2);
  f16* Qin  = (f16*)alloc((size_t)NTOK * EE * 2);
  f16* X    = (f16*)alloc((size_t)NTOK * EE * 2);
  f16* Qs   = (f16*)alloc((size_t)NTOK * EE * 2);
  f16* Kmat = (f16*)alloc((size_t)NTOK * EE * 2);
  f16* VtT  = (f16*)alloc((size_t)EE * NTOK * 2);   // V^T [EE][NTOK]
  f16* Win6 = (f16*)alloc((size_t)EE * 512 * 2);
  f16* Wq6  = (f16*)alloc((size_t)EE * EE * 2);
  f16* Wk6  = (f16*)alloc((size_t)EE * EE * 2);
  f16* Wv6  = (f16*)alloc((size_t)EE * EE * 2);

  // weights straight f32 -> f16 (row-major [N][K] is exactly the B layout we load)
  cvt_f16<<<(EE * 512 + 255) / 256, 256, 0, stream>>>(W_in, (long)EE * 512, Win6);
  cvt_f16<<<(EE * EE + 255) / 256, 256, 0, stream>>>(Wq, (long)EE * EE, Wq6);
  cvt_f16<<<(EE * EE + 255) / 256, 256, 0, stream>>>(Wk, (long)EE * EE, Wk6);
  cvt_f16<<<(EE * EE + 255) / 256, 256, 0, stream>>>(Wv, (long)EE * EE, Wv6);

  gather_build<<<NTOK, 256, 0, stream>>>(q, r, nxt, emb, Ain, Qin);

  const int Mt = NTOK / 16, Nt = EE / 16;           // 1024 x 16 tiles
  const int nblk = (Mt * Nt) / 8;                   // 8 waves/block
  // X = relu(Ain x W_in^T + b_in)
  gemm_f16<<<nblk, 256, 0, stream>>>(Ain, 512, Win6, 512, b_in, X, EE, Mt, Nt, 512, 1.0f, 1, 0);
  // Q = (Qin x Wq^T + bq) / sqrt(E)
  gemm_f16<<<nblk, 256, 0, stream>>>(Qin, EE, Wq6, EE, bq, Qs, EE, Mt, Nt, EE, 0.0625f, 0, 0);
  // K = X x Wk^T + bk
  gemm_f16<<<nblk, 256, 0, stream>>>(X, EE, Wk6, EE, bk, Kmat, EE, Mt, Nt, EE, 1.0f, 0, 0);
  // V^T = (X x Wv^T + bv)^T   (transposed epilogue, v8h stores)
  gemm_f16<<<nblk, 256, 0, stream>>>(X, EE, Wv6, EE, bv, VtT, NTOK, Mt, Nt, EE, 1.0f, 0, 1);

  float* logits = (float*)d_out;                    // [B,L,1]
  float* pout   = (float*)d_out + NTOK;             // [B,1,L,L]
  attn_fused<<<BB * 64, 256, SM_BYTES, stream>>>(Qs, Kmat, VtT, rel, ts, Wout, bout,
                                                 l1, l2, logits, pout);
}
